// MultiHeadGraphAttentionLayer_55482387529960
// MI455X (gfx1250) — matrix operand order repaired and checked by reading, not compiled, not run on hardware
//
#include <hip/hip_runtime.h>
#include <hip/hip_bf16.h>

#define GN 4096
#define IN_DIM 512
#define OUT_DIM 256
#define NH 4
#define HD 64
#define NEG_SLOPE 0.2f
#define LN_EPS 1e-5f

typedef __attribute__((ext_vector_type(2)))  float    v2f;
typedef __attribute__((ext_vector_type(8)))  float    v8f;
typedef __attribute__((ext_vector_type(16))) _Float16 v16h;

typedef unsigned int u32;
typedef unsigned long long u64;
typedef __attribute__((ext_vector_type(4))) u32 u32x4;
typedef __attribute__((ext_vector_type(4))) int i32x4;
typedef __attribute__((ext_vector_type(8))) int i32x8;

// Low 32 bits of a generic pointer to LDS == byte offset in LDS.
__device__ __forceinline__ u32 lds_off(const void* p) {
    return (u32)(size_t)p;
}

// Issue one TDM 2D tile load (Global -> LDS) per CDNA5 D# descriptor layout.
//  tile0      : row length (elements, 4B each)
//  tile1      : number of rows (1 => effectively 1-D)
//  stride0    : tensor dim-0 stride (elements)
//  pad_code   : pad_interval code (5 => every 64 DWORDs, 6 => every 128 DWORDs)
//  pad_amt    : pad_amount code (1 => 2 DWORDs of LDS padding)
__device__ __forceinline__ void tdm_load_2d(u32 lds_addr, const void* gptr,
                                            u32 tile0, u32 tile1, u32 stride0,
                                            u32 pad_code, u32 pad_amt)
{
    u64 ga = (u64)(size_t)gptr;
    const u32 TD = 1u << 20;            // generous tensor dims: no OOB clamping
    u32x4 g0;
    g0[0] = 1u;                                              // count=1, user mode
    g0[1] = lds_addr;                                        // lds_addr (bytes)
    g0[2] = (u32)ga;                                         // global_addr[31:0]
    g0[3] = (u32)((ga >> 32) & 0x1FFFFFFu) | (2u << 30);     // addr[56:32] | type=2
    i32x8 g1;
    g1[0] = (int)((2u << 16) | (1u << 20) |                  // data_size=4B, pad_en
                  (pad_code << 22) | (pad_amt << 25));
    g1[1] = 0;                                               // no atomic barrier
    g1[2] = (int)((TD >> 16) & 0xFFFFu);                     // tensor_dim0 hi
    g1[3] = (int)(((TD >> 16) & 0xFFFFu) | (tile0 << 16));   // tensor_dim1 hi|tile0
    g1[4] = (int)tile1;                                      // tile1 | tile2=0
    g1[5] = (int)stride0;                                    // dim0 stride lo
    g1[6] = 0;
    g1[7] = 0;
    i32x4 gz4 = {0, 0, 0, 0};
    i32x8 gz8 = {0, 0, 0, 0, 0, 0, 0, 0};
    // 6-arg toolchain variant: (g0, g1, g2, g3, <extra group>, cpol)
    __builtin_amdgcn_tensor_load_to_lds(g0, g1, gz4, gz4, gz8, 0);
}

// ---------------------------------------------------------------------------
// Kernel 1: per-head projection Wx[h][n][d] = x[n][:] . W[h][:][d] (fp32 WMMA).
// Also writes fp16 transposed copy WxhT[h][d][n] (attention V-operand).
// 128 threads (4 waves); wave w owns output cols [16w,16w+16).
// K staged through LDS in chunks of 128 by the Tensor Data Mover; TDM pad
// fields insert +2 DWORDs per row, giving the bank-conflict-free pitches
// xs[16][130] and ws[128][66] with a single instruction per tile.
// ---------------------------------------------------------------------------
__global__ __launch_bounds__(128) void gat_proj_kernel(
    const float* __restrict__ x, const float* __restrict__ W,
    float* __restrict__ Wx, _Float16* __restrict__ WxhT)
{
    __shared__ float xs[16 * 130];   // x tile, pitch 130 (TDM-padded)
    __shared__ float ws[128 * 66];   // W tile, pitch 66  (TDM-padded)

    const int tid  = threadIdx.x;
    const int wave = tid >> 5;
    const int lane = tid & 31;
    const int hlf  = lane >> 4;
    const int l16  = lane & 15;
    const int i0   = blockIdx.x * 16;
    const int h    = blockIdx.y;
    const u32 xs_o = lds_off(xs);
    const u32 ws_o = lds_off(ws);

    v8f acc = {};
    for (int kc = 0; kc < IN_DIM; kc += 128) {
        __syncthreads();                      // previous chunk fully consumed
        if (wave == 0) {
            // x rows i0..i0+15, cols kc..kc+127 (row stride IN_DIM)
            tdm_load_2d(xs_o, x + (size_t)i0 * IN_DIM + kc, 128, 16, IN_DIM, 6, 1);
            // W chunk: 128 K-rows x 64 cols, contiguous 8192 floats
            tdm_load_2d(ws_o, W + ((size_t)h * IN_DIM + kc) * HD, 8192, 1, 8192, 5, 1);
            __builtin_amdgcn_s_wait_tensorcnt(0);
        }
        __syncthreads();
        // A 16x4 f32 frag: lane<16 row=l16 K={k,k+1}; lane>=16 K={k+2,k+3}
        // B 4x16  f32 frag: same K split, col = 16*wave + l16
        const int colw = wave * 16 + l16;
        #pragma unroll 8
        for (int kk = 0; kk < 128; kk += 4) {
            int ko = kk + 2 * hlf;
            v2f av = *(const v2f*)&xs[l16 * 130 + ko];
            v2f bv = {ws[ko * 66 + colw], ws[(ko + 1) * 66 + colw]};
            acc = __builtin_amdgcn_wmma_f32_16x16x4_f32(
                false, av, false, bv, (short)0, acc, false, false);
        }
    }
    // C/D layout: VGPR r -> row r + 8*hlf, col l16
    const int col = wave * 16 + l16;
    #pragma unroll
    for (int r = 0; r < 8; ++r) {
        int row = i0 + r + 8 * hlf;
        float v = acc[r];
        Wx[((size_t)h * GN + row) * HD + col]   = v;
        WxhT[((size_t)h * HD + col) * GN + row] = (_Float16)v;   // contiguous -> b128
    }
}

// ---------------------------------------------------------------------------
// Kernel 1b: a_src[h][n] = Wx[h][n][:].a[h][:64], a_dst with a[h][64:]
// ---------------------------------------------------------------------------
__global__ __launch_bounds__(128) void gat_attvec_kernel(
    const float* __restrict__ Wx, const float* __restrict__ a,
    float* __restrict__ a_src, float* __restrict__ a_dst)
{
    int gid = blockIdx.x * 128 + threadIdx.x;   // 0 .. NH*GN-1
    int h = gid >> 12;
    const float4* wv = (const float4*)(Wx + (size_t)gid * HD);
    const float4* as = (const float4*)(a + h * 2 * HD);
    const float4* ad = (const float4*)(a + h * 2 * HD + HD);
    float s = 0.f, d = 0.f;
    #pragma unroll
    for (int i = 0; i < 16; ++i) {
        float4 w = wv[i], va = as[i], vd = ad[i];
        s += w.x * va.x + w.y * va.y + w.z * va.z + w.w * va.w;
        d += w.x * vd.x + w.y * vd.y + w.z * vd.z + w.w * vd.w;
    }
    a_src[gid] = s;
    a_dst[gid] = d;
}

// ---------------------------------------------------------------------------
// Kernel 2: fused masked-softmax attention + ELU + LayerNorm.
// One block per 32-row tile; 8 waves = 2 row-groups x 4 heads, so each
// WxhT B-fragment is shared by two waves through the WGP cache (halves the
// dominant L2 stream vs 16-row blocks). Flash-style online softmax; the
// probability tile is built directly in WMMA A-fragment layout and fed to
// v_wmma_f32_16x16x32_f16 without touching LDS.
// ---------------------------------------------------------------------------
__global__ __launch_bounds__(256) void gat_attn_kernel(
    const int* __restrict__ adj, const _Float16* __restrict__ WxhT,
    const float* __restrict__ a_src, const float* __restrict__ a_dst,
    const float* __restrict__ gamma, const float* __restrict__ beta,
    float* __restrict__ out)
{
    __shared__ float outbuf[32][260];   // padded vs 64-bank LDS
    __shared__ float red[32][8][2];
    __shared__ float mu_s[32], rs_s[32];

    const int tid  = threadIdx.x;
    const int wave = tid >> 5;
    const int h    = wave & 3;          // head
    const int rg   = wave >> 2;         // row-group (0/1)
    const int lane = tid & 31;
    const int hlf  = lane >> 4;
    const int l16  = lane & 15;
    const int bi0  = blockIdx.x * 32;
    const int i0   = bi0 + rg * 16;
    const int i_row = i0 + l16;         // softmax-state row for this lane

    const float asrc = a_src[h * GN + i_row];
    const float* adst_h = a_dst + h * GN;
    const int* adj_row  = adj + (size_t)i_row * GN;
    const _Float16* Wh  = WxhT + (size_t)h * HD * GN;

    float m = -1e30f, lsum = 0.f;
    v8f O0 = {}, O1 = {}, O2 = {}, O3 = {};

    for (int j0 = 0; j0 < GN; j0 += 32) {
        if (j0 + 96 < GN)                                  // stream adj ahead
            __builtin_prefetch(adj_row + j0 + 64, 0, 1);

        // ---- scores in A-fragment order -------------------------------
        // lane<16:  K = {j0..j0+15}
        // lane>=16: K = {j0+16..j0+31}
        float s[16];
        #pragma unroll
        for (int rr = 0; rr < 2; ++rr) {
            const int jb = j0 + rr * 8 + 16 * hlf;
            float4 d0 = *(const float4*)(adst_h + jb);
            float4 d1 = *(const float4*)(adst_h + jb + 4);
            int4   m0 = *(const int4*)(adj_row + jb);
            int4   m1 = *(const int4*)(adj_row + jb + 4);
            float dv[8] = {d0.x, d0.y, d0.z, d0.w, d1.x, d1.y, d1.z, d1.w};
            int   mv[8] = {m0.x, m0.y, m0.z, m0.w, m1.x, m1.y, m1.z, m1.w};
            #pragma unroll
            for (int e = 0; e < 8; ++e) {
                float sc = asrc + dv[e];
                sc = fmaxf(sc, NEG_SLOPE * sc);            // leaky relu
                bool edge = (mv[e] > 0) || (i_row == jb + e);
                s[rr * 8 + e] = edge ? sc : -1e30f;
            }
        }
        // ---- online softmax (row state at lanes l16 / l16+16) ---------
        float mloc = s[0];
        #pragma unroll
        for (int e = 1; e < 16; ++e) mloc = fmaxf(mloc, s[e]);
        mloc = fmaxf(mloc, __shfl_xor(mloc, 16, 32));
        float mnew  = fmaxf(m, mloc);
        float scale = __expf(m - mnew);
        m = mnew;
        float rs = 0.f;
        v16h pa;
        #pragma unroll
        for (int e = 0; e < 16; ++e) {
            float p = __expf(s[e] - m);
            rs += p;
            pa[e] = (_Float16)p;
        }
        rs += __shfl_xor(rs, 16, 32);
        lsum = lsum * scale + rs;
        // ---- rescale accumulators (rows r + 8*hlf) --------------------
        float scr[8];
        #pragma unroll
        for (int r = 0; r < 8; ++r) scr[r] = __shfl(scale, 8 * hlf + r, 32);
        #pragma unroll
        for (int r = 0; r < 8; ++r) {
            O0[r] *= scr[r]; O1[r] *= scr[r];
            O2[r] *= scr[r]; O3[r] *= scr[r];
        }
        // ---- PV: B 32x16 f16 frag = 16 contiguous halfs along j -------
        const int jb = j0 + 16 * hlf;
        v16h b0 = *(const v16h*)(Wh + (size_t)(l16)      * GN + jb);
        O0 = __builtin_amdgcn_wmma_f32_16x16x32_f16(false, pa, false, b0, (short)0, O0, false, false);
        v16h b1 = *(const v16h*)(Wh + (size_t)(16 + l16) * GN + jb);
        O1 = __builtin_amdgcn_wmma_f32_16x16x32_f16(false, pa, false, b1, (short)0, O1, false, false);
        v16h b2 = *(const v16h*)(Wh + (size_t)(32 + l16) * GN + jb);
        O2 = __builtin_amdgcn_wmma_f32_16x16x32_f16(false, pa, false, b2, (short)0, O2, false, false);
        v16h b3 = *(const v16h*)(Wh + (size_t)(48 + l16) * GN + jb);
        O3 = __builtin_amdgcn_wmma_f32_16x16x32_f16(false, pa, false, b3, (short)0, O3, false, false);
    }

    // ---- finalize: O /= l, ELU, stage into LDS ------------------------
    float li[8];
    #pragma unroll
    for (int r = 0; r < 8; ++r) li[r] = __shfl(lsum, 8 * hlf + r, 32);
    #pragma unroll
    for (int r = 0; r < 8; ++r) {
        int row = r + 8 * hlf;
        float inv = 1.f / li[r];
        float vals[4] = {O0[r], O1[r], O2[r], O3[r]};
        #pragma unroll
        for (int nt = 0; nt < 4; ++nt) {
            float v = vals[nt] * inv;
            v = v > 0.f ? v : (__expf(v) - 1.f);           // ELU
            outbuf[rg * 16 + row][h * HD + nt * 16 + l16] = v;
        }
    }
    __syncthreads();

    // ---- LayerNorm over 256 cols (8 threads per row, 32 rows) ---------
    {
        int row = tid >> 3, sub = tid & 7;
        float s1 = 0.f, s2 = 0.f;
        #pragma unroll 8
        for (int k = 0; k < 32; ++k) {
            float v = outbuf[row][sub * 32 + k];
            s1 += v; s2 += v * v;
        }
        red[row][sub][0] = s1; red[row][sub][1] = s2;
    }
    __syncthreads();
    if ((tid & 7) == 0) {
        int row = tid >> 3;
        float s1 = 0.f, s2 = 0.f;
        #pragma unroll
        for (int k = 0; k < 8; ++k) { s1 += red[row][k][0]; s2 += red[row][k][1]; }
        float mu  = s1 * (1.0f / OUT_DIM);
        float var = s2 * (1.0f / OUT_DIM) - mu * mu;
        mu_s[row] = mu;
        rs_s[row] = rsqrtf(var + LN_EPS);
    }
    __syncthreads();
    #pragma unroll 4
    for (int e = 0; e < 32; ++e) {
        int flat = e * 256 + tid;
        int row = flat >> 8, col = flat & 255;
        float v = outbuf[row][col];
        out[(size_t)(bi0 + row) * OUT_DIM + col] =
            (v - mu_s[row]) * rs_s[row] * gamma[col] + beta[col];
    }
}

// ---------------------------------------------------------------------------
extern "C" void kernel_launch(void* const* d_in, const int* in_sizes, int n_in,
                              void* d_out, int out_size, void* d_ws, size_t ws_size,
                              hipStream_t stream) {
    const float* x     = (const float*)d_in[0];
    const int*   adj   = (const int*)  d_in[1];
    const float* W     = (const float*)d_in[2];
    const float* a     = (const float*)d_in[3];
    const float* gamma = (const float*)d_in[4];
    const float* beta  = (const float*)d_in[5];
    float* out = (float*)d_out;

    // workspace: Wx fp32 (4MB) | WxhT fp16 (2MB) | a_src (64KB) | a_dst (64KB)
    char* ws = (char*)d_ws;
    float*    Wx    = (float*)ws;
    _Float16* WxhT  = (_Float16*)(ws + (4u << 20));
    float*    a_src = (float*)(ws + (6u << 20));
    float*    a_dst = (float*)(ws + (6u << 20) + (64u << 10));

    gat_proj_kernel<<<dim3(GN / 16, NH), 128, 0, stream>>>(x, W, Wx, WxhT);
    gat_attvec_kernel<<<dim3(NH * GN / 128), 128, 0, stream>>>(Wx, a, a_src, a_dst);
    gat_attn_kernel<<<dim3(GN / 32), 256, 0, stream>>>(adj, WxhT, a_src, a_dst,
                                                       gamma, beta, out);
}
